// CapsuleNet_36472862277928
// MI455X (gfx1250) — matrix operand order repaired and checked by reading, not compiled, not run on hardware
//
#include <hip/hip_runtime.h>

typedef __attribute__((ext_vector_type(16))) _Float16 v16h;
typedef __attribute__((ext_vector_type(8)))  _Float16 v8h;
typedef __attribute__((ext_vector_type(8)))  float    v8f;

#define WMMA_F16(a, b, c) \
  __builtin_amdgcn_wmma_f32_16x16x32_f16(false, (a), false, (b), (short)0, (c), false, false)

// Issue the CDNA5 async global->LDS copy path (ASYNCcnt-tracked) for the
// prim-GEMM weight staging. Set to 0 to fall back to plain generic stores.
#define USE_ASYNC_LDS 1

// ---- fragment loaders (CDNA5 wave32 WMMA layouts) ----
// A 16x32 f16: lane row = m0+(l&15); halves[0..15] = contiguous K at 16*(l>=16)
__device__ __forceinline__ v16h load_a_frag(const _Float16* rowk, int lane) {
  const _Float16* p = rowk + ((lane >> 4) << 4);
  union { v16h v; v8h h[2]; } u;
  u.h[0] = *(const v8h*)(p);
  u.h[1] = *(const v8h*)(p + 8);
  return u.v;
}
// B 32x16 f16: lane col = n0+(l&15); halves[0..15] = contiguous K at 16*(l>=16)
__device__ __forceinline__ v16h load_b_frag(const _Float16* colk, int lane) {
  const _Float16* p = colk + ((lane >> 4) << 4);
  union { v16h v; v8h h[2]; } u;
  u.h[0] = *(const v8h*)(p);
  u.h[1] = *(const v8h*)(p + 8);
  return u.v;
}
// B frag out of an LDS-staged [64 cols][32 k] tile
__device__ __forceinline__ v16h load_b_frag_lds(const _Float16* Bs, int colbase, int lane) {
  const _Float16* p = Bs + (size_t)(colbase + (lane & 15)) * 32 + ((lane >> 4) << 4);
  union { v16h v; v8h h[2]; } u;
  u.h[0] = *(const v8h*)(p);
  u.h[1] = *(const v8h*)(p + 8);
  return u.v;
}

#if USE_ASYNC_LDS
typedef __attribute__((address_space(3))) void as3void;
__device__ __forceinline__ void async_copy16_to_lds(_Float16* lds_dst, const _Float16* gsrc) {
  unsigned ldsOff = (unsigned)(size_t)(as3void*)lds_dst;  // 32-bit LDS byte offset
  unsigned long long gaddr = (unsigned long long)(size_t)gsrc;
  asm volatile("global_load_async_to_lds_b128 %0, %1, off"
               :: "v"(ldsOff), "v"(gaddr) : "memory");
}
__device__ __forceinline__ void wait_async0() {
  asm volatile("s_wait_asynccnt 0x0" ::: "memory");
}
#endif

// ================= prep kernels =================

// im2col for conv1: A1[m=(b,oy,ox)][k=ky*9+kx, padded to 96] f16, M=102400
__global__ void k_im2col_conv1(const float* __restrict__ x, _Float16* __restrict__ A1) {
  int tid = blockIdx.x * 256 + threadIdx.x;
  if (tid >= 102400 * 96) return;
  int k = tid % 96;
  int m = tid / 96;
  int b = m / 400, s = m % 400, oy = s / 20, ox = s % 20;
  float v = 0.f;
  if (k < 81) {
    int ky = k / 9, kx = k % 9;
    v = x[b * 784 + (oy + ky) * 28 + (ox + kx)];
  }
  A1[tid] = (_Float16)v;
}

// conv1 weights -> [O=256][96] f16 (K padded 81 -> 96)
__global__ void k_prep_w1(const float* __restrict__ w, _Float16* __restrict__ dst) {
  int tid = blockIdx.x * 256 + threadIdx.x;
  if (tid >= 256 * 96) return;
  int o = tid / 96, k = tid % 96;
  dst[tid] = (k < 81) ? (_Float16)w[o * 81 + k] : (_Float16)0.f;
}

// prim weights [O=256][C=256][9][9] -> [pos=81][O=256][C=256] f16
__global__ void k_prep_w2(const float* __restrict__ w, _Float16* __restrict__ dst) {
  int tid = blockIdx.x * 256 + threadIdx.x;
  if (tid >= 81 * 256 * 256) return;
  int c = tid & 255;
  int o = (tid >> 8) & 255;
  int pos = tid >> 16;
  dst[tid] = (_Float16)w[((size_t)(o * 256 + c)) * 81 + pos];
}

// f32 [K][N] -> f16 [N][K] (decoder weight transposes)
__global__ void k_transpose_f16(const float* __restrict__ src, _Float16* __restrict__ dst,
                                int K, int N) {
  int tid = blockIdx.x * 256 + threadIdx.x;
  if (tid >= K * N) return;
  int n = tid / K, k = tid % K;
  dst[(size_t)n * K + k] = (_Float16)src[(size_t)k * N + n];
}

// ================= WMMA GEMM kernels =================

// conv1: C[M=102400, N=256] = A1[M,96] x W1[N,96]^T ; 16x64 tile per wave
// (1 A-frag amortized over 4 B-frags). Epilogue bias+relu -> NHWC f16.
__global__ void k_conv1_gemm(const _Float16* __restrict__ A, const _Float16* __restrict__ Bt,
                             const float* __restrict__ bias, _Float16* __restrict__ h_nhwc) {
  int lane = threadIdx.x;
  int tile = blockIdx.x * 8 + threadIdx.y;  // 25600 wave tiles
  int tm = tile >> 2, tn = tile & 3;
  int m0 = tm * 16, n0 = tn * 64;
  const _Float16* arow = A + (size_t)(m0 + (lane & 15)) * 96;
  const _Float16* brow = Bt + (size_t)(n0 + (lane & 15)) * 96;
  v8f acc[4] = {};
#pragma unroll
  for (int k0 = 0; k0 < 96; k0 += 32) {
    v16h af = load_a_frag(arow + k0, lane);
#pragma unroll
    for (int j = 0; j < 4; ++j) {
      v16h bf = load_b_frag(brow + (size_t)j * 16 * 96 + k0, lane);
      acc[j] = WMMA_F16(af, bf, acc[j]);
    }
  }
  int mb = m0 + ((lane >> 4) << 3);
#pragma unroll
  for (int j = 0; j < 4; ++j) {
    int n = n0 + j * 16 + (lane & 15);
    float bn = bias[n];
#pragma unroll
    for (int v = 0; v < 8; ++v) {
      float val = acc[j][v] + bn;
      val = val > 0.f ? val : 0.f;
      h_nhwc[(size_t)(mb + v) * 256 + n] = (_Float16)val;
    }
  }
}

// prim caps: implicit GEMM, C[M=9216=(b,oy,ox), N=256] over K = 81 taps x 256 ch.
// Block tile 128x64 (8 waves = 4m x 2n of 32x32 wave tiles). Weight k-slice
// (64 cols x 32 k, 4KB) staged into LDS via async global->LDS copies.
// Epilogue bias + scatter into p[b][r=ch32*36+s][g=o>>5] f32.
__global__ void k_prim_gemm(const _Float16* __restrict__ h_nhwc, const _Float16* __restrict__ W2,
                            const float* __restrict__ bias, float* __restrict__ p) {
  __shared__ __align__(16) _Float16 Bs[64 * 32];
  int lane = threadIdx.x;
  int w = threadIdx.y;               // 0..7
  int waveM = w >> 1, waveN = w & 1; // 4 x 2
  int mb_idx = blockIdx.x % 72;
  int nb_idx = blockIdx.x / 72;
  int mblk = mb_idx * 128, nblk = nb_idx * 64;
  int m0 = mblk + waveM * 32, n0 = nblk + waveN * 32;

  // per-lane A row bases for the two 16-row groups
  int r0 = m0 + (lane & 15), r1 = r0 + 16;
  int b0 = r0 / 36, s0 = r0 % 36;
  int b1 = r1 / 36, s1 = r1 % 36;
  const _Float16* abase0 =
      h_nhwc + ((size_t)b0 * 400 + (size_t)(s0 / 6) * 40 + (size_t)(s0 % 6) * 2) * 256;
  const _Float16* abase1 =
      h_nhwc + ((size_t)b1 * 400 + (size_t)(s1 / 6) * 40 + (size_t)(s1 % 6) * 2) * 256;

  // staging assignment: 256 threads x 16B = 4KB tile
  int t256 = w * 32 + lane;
  int srow = t256 >> 2;              // 0..63 (column of B = output channel)
  int sseg = (t256 & 3) << 3;        // half-element offset 0/8/16/24
  const _Float16* gstage_base = W2 + ((size_t)(nblk + srow)) * 256 + sseg;
  _Float16* lstage = &Bs[srow * 32 + sseg];

  v8f acc[2][2] = {};
  for (int pos = 0; pos < 81; ++pos) {
    int ky = pos / 9, kx = pos % 9;
    int poff = (ky * 20 + kx) * 256;
    const _Float16* gst = gstage_base + (size_t)pos * 256 * 256;
#pragma unroll 1
    for (int c0 = 0; c0 < 256; c0 += 32) {
      __syncthreads();  // previous step's LDS reads done
#if USE_ASYNC_LDS
      async_copy16_to_lds(lstage, gst + c0);
      wait_async0();
#else
      *(v8h*)lstage = *(const v8h*)(gst + c0);
#endif
      __syncthreads();  // tile staged by all waves

      v16h a0 = load_a_frag(abase0 + poff + c0, lane);
      v16h a1 = load_a_frag(abase1 + poff + c0, lane);
      v16h bf0 = load_b_frag_lds(Bs, waveN * 32, lane);
      v16h bf1 = load_b_frag_lds(Bs, waveN * 32 + 16, lane);
      acc[0][0] = WMMA_F16(a0, bf0, acc[0][0]);
      acc[0][1] = WMMA_F16(a0, bf1, acc[0][1]);
      acc[1][0] = WMMA_F16(a1, bf0, acc[1][0]);
      acc[1][1] = WMMA_F16(a1, bf1, acc[1][1]);
    }
  }

#pragma unroll
  for (int mi = 0; mi < 2; ++mi)
#pragma unroll
    for (int ni = 0; ni < 2; ++ni) {
      int o = n0 + ni * 16 + (lane & 15);
      float bo = bias[o];
      int mb = m0 + mi * 16 + ((lane >> 4) << 3);
#pragma unroll
      for (int v = 0; v < 8; ++v) {
        int mm = mb + v;
        int bb = mm / 36, ss = mm % 36;
        p[(size_t)bb * 9216 + (o & 31) * 288 + ss * 8 + (o >> 5)] = acc[mi][ni][v] + bo;
      }
    }
}

// decoder FC: C[M=256,N] = A[M,K] x Wt[N,K]^T ; mode 0: bias+relu -> f16 [M,N];
// mode 1: bias+sigmoid -> f32 [M,N]
__global__ void k_fc_gemm(const _Float16* __restrict__ A, const _Float16* __restrict__ Bt,
                          const float* __restrict__ bias, int N, int K, int mode,
                          _Float16* __restrict__ out_h, float* __restrict__ out_f) {
  int lane = threadIdx.x;
  int tile = blockIdx.x * 8 + threadIdx.y;
  int ntn = N >> 4;
  int tm = tile / ntn, tn = tile % ntn;
  int m0 = tm * 16, n0 = tn * 16;
  const _Float16* arow = A + (size_t)(m0 + (lane & 15)) * K;
  const _Float16* brow = Bt + (size_t)(n0 + (lane & 15)) * K;
  v8f acc = {};
  for (int k0 = 0; k0 < K; k0 += 32) {
    v16h af = load_a_frag(arow + k0, lane);
    v16h bf = load_b_frag(brow + k0, lane);
    acc = WMMA_F16(af, bf, acc);
  }
  int n = n0 + (lane & 15);
  int mb = m0 + ((lane >> 4) << 3);
  float bn = bias[n];
#pragma unroll
  for (int v = 0; v < 8; ++v) {
    float val = acc[v] + bn;
    if (mode == 0) {
      val = val > 0.f ? val : 0.f;
      out_h[(size_t)(mb + v) * N + n] = (_Float16)val;
    } else {
      val = 1.f / (1.f + __expf(-val));
      out_f[(size_t)(mb + v) * N + n] = val;
    }
  }
}

// ================= f32 VALU kernels =================

// squash the 8-dim capsule vectors in place: p[B*1152][8]
__global__ void k_squash8(float* __restrict__ p) {
  int tid = blockIdx.x * 256 + threadIdx.x;
  if (tid >= 256 * 1152) return;
  float* q = p + (size_t)tid * 8;
  float sq = 0.f;
#pragma unroll
  for (int i = 0; i < 8; ++i) sq += q[i] * q[i];
  float scale = (sq / (1.f + sq)) / sqrtf(sq);
#pragma unroll
  for (int i = 0; i < 8; ++i) q[i] *= scale;
}

// priors[c][b][r][o] = sum_i p[b][r][i] * route_w[c][r][i][o]
__global__ void k_priors(const float* __restrict__ p, const float* __restrict__ rw,
                         float* __restrict__ priors) {
  size_t tid = (size_t)blockIdx.x * 256 + threadIdx.x;
  if (tid >= 47185920ull) return;
  int o = (int)(tid & 15);
  size_t t = tid >> 4;
  int r = (int)(t % 1152); t /= 1152;
  int b = (int)(t % 256);
  int c = (int)(t / 256);
  const float* pp = p + ((size_t)b * 1152 + r) * 8;
  const float* ww = rw + (((size_t)c * 1152 + r) * 8) * 16 + o;
  float s = 0.f;
#pragma unroll
  for (int i = 0; i < 8; ++i) s += pp[i] * ww[i * 16];
  priors[tid] = s;
}

// dynamic routing, 3 iterations; one block per (c,b); 128 threads x 9 route nodes
__global__ void k_routing(const float* __restrict__ priors, float* __restrict__ vout) {
  int c = blockIdx.x / 256;
  int b = blockIdx.x % 256;
  const float* P = priors + ((size_t)(c * 256 + b)) * 1152 * 16;
  int t = threadIdx.x;
  __shared__ float red[128];
  __shared__ float smem[16 * 128];
  __shared__ float sv[16];
  __shared__ float outv[16];
  float logit[9];
#pragma unroll
  for (int j = 0; j < 9; ++j) logit[j] = 0.f;
  for (int iter = 0; iter < 3; ++iter) {
    float mx = logit[0];
#pragma unroll
    for (int j = 1; j < 9; ++j) mx = fmaxf(mx, logit[j]);
    red[t] = mx; __syncthreads();
    for (int off = 64; off > 0; off >>= 1) {
      if (t < off) red[t] = fmaxf(red[t], red[t + off]);
      __syncthreads();
    }
    float gmax = red[0]; __syncthreads();
    float prob[9];
    float esum = 0.f;
#pragma unroll
    for (int j = 0; j < 9; ++j) { prob[j] = __expf(logit[j] - gmax); esum += prob[j]; }
    red[t] = esum; __syncthreads();
    for (int off = 64; off > 0; off >>= 1) {
      if (t < off) red[t] += red[t + off];
      __syncthreads();
    }
    float inv = 1.f / red[0]; __syncthreads();
    float part[16];
#pragma unroll
    for (int o = 0; o < 16; ++o) part[o] = 0.f;
    for (int j = 0; j < 9; ++j) {
      const float* pr = P + (size_t)(t * 9 + j) * 16;
      float wgt = prob[j] * inv;
#pragma unroll
      for (int o = 0; o < 16; ++o) part[o] += wgt * pr[o];
    }
#pragma unroll
    for (int o = 0; o < 16; ++o) smem[o * 128 + t] = part[o];
    __syncthreads();
    if (t < 16) {
      float s = 0.f;
      for (int k = 0; k < 128; ++k) s += smem[t * 128 + k];
      sv[t] = s;
    }
    __syncthreads();
    if (t == 0) {
      float sq = 0.f;
      for (int o = 0; o < 16; ++o) sq += sv[o] * sv[o];
      float scale = (sq / (1.f + sq)) / sqrtf(sq);
      for (int o = 0; o < 16; ++o) outv[o] = sv[o] * scale;
    }
    __syncthreads();
    if (iter < 2) {
      for (int j = 0; j < 9; ++j) {
        const float* pr = P + (size_t)(t * 9 + j) * 16;
        float d = 0.f;
#pragma unroll
        for (int o = 0; o < 16; ++o) d += pr[o] * outv[o];
        logit[j] += d;
      }
    }
  }
  if (t < 16) vout[((size_t)b * 10 + c) * 16 + t] = outv[t];
}

// class probabilities (softmax over capsule norms) + argmax mask -> masked f16 [B][160]
__global__ void k_classes(const float* __restrict__ vout, float* __restrict__ classes_out,
                          _Float16* __restrict__ masked) {
  int b = blockIdx.x * blockDim.x + threadIdx.x;
  if (b >= 256) return;
  float norm[10];
  float mx = -1e30f;
  for (int c = 0; c < 10; ++c) {
    float sq = 0.f;
    for (int o = 0; o < 16; ++o) {
      float t = vout[((size_t)b * 10 + c) * 16 + o];
      sq += t * t;
    }
    norm[c] = sqrtf(sq);
    mx = fmaxf(mx, norm[c]);
  }
  float e[10];
  float es = 0.f;
  for (int c = 0; c < 10; ++c) { e[c] = __expf(norm[c] - mx); es += e[c]; }
  int am = 0;
  float best = -1.f;
  for (int c = 0; c < 10; ++c) {
    float cl = e[c] / es;
    classes_out[b * 10 + c] = cl;
    if (cl > best) { best = cl; am = c; }
  }
  for (int c = 0; c < 10; ++c)
    for (int o = 0; o < 16; ++o) {
      float v = (c == am) ? vout[((size_t)b * 10 + c) * 16 + o] : 0.f;
      masked[(size_t)b * 160 + c * 16 + o] = (_Float16)v;
    }
}

// ================= host launcher =================

extern "C" void kernel_launch(void* const* d_in, const int* in_sizes, int n_in,
                              void* d_out, int out_size, void* d_ws, size_t ws_size,
                              hipStream_t stream) {
  const float* x       = (const float*)d_in[0];
  const float* conv1_w = (const float*)d_in[1];
  const float* conv1_b = (const float*)d_in[2];
  const float* prim_w  = (const float*)d_in[3];
  const float* prim_b  = (const float*)d_in[4];
  const float* route_w = (const float*)d_in[5];
  const float* dec_w1  = (const float*)d_in[6];
  const float* dec_b1  = (const float*)d_in[7];
  const float* dec_w2  = (const float*)d_in[8];
  const float* dec_b2  = (const float*)d_in[9];
  const float* dec_w3  = (const float*)d_in[10];
  const float* dec_b3  = (const float*)d_in[11];

  float* out_classes = (float*)d_out;           // [256,10]
  float* out_recon   = (float*)d_out + 2560;    // [256,784]

  char* ws = (char*)d_ws;
  auto carve = [&](size_t bytes) -> char* {
    char* p = ws;
    ws += (bytes + 255) & ~((size_t)255);
    return p;
  };
  _Float16* A1     = (_Float16*)carve((size_t)102400 * 96 * 2);   // conv1 im2col
  _Float16* w1h    = (_Float16*)carve((size_t)256 * 96 * 2);
  _Float16* h_nhwc = (_Float16*)carve((size_t)256 * 400 * 256 * 2);
  _Float16* w2h    = (_Float16*)carve((size_t)81 * 256 * 256 * 2);
  float*    p      = (float*)   carve((size_t)256 * 1152 * 8 * 4);
  float*    priors = (float*)   carve((size_t)10 * 256 * 1152 * 16 * 4);
  float*    vout   = (float*)   carve((size_t)256 * 10 * 16 * 4);
  _Float16* masked = (_Float16*)carve((size_t)256 * 160 * 2);
  _Float16* d1t    = (_Float16*)carve((size_t)512 * 160 * 2);
  _Float16* h1     = (_Float16*)carve((size_t)256 * 512 * 2);
  _Float16* d2t    = (_Float16*)carve((size_t)1024 * 512 * 2);
  _Float16* h2     = (_Float16*)carve((size_t)256 * 1024 * 2);
  _Float16* d3t    = (_Float16*)carve((size_t)784 * 1024 * 2);

  dim3 wblk(32, 8);

  // prep
  k_im2col_conv1<<<38400, 256, 0, stream>>>(x, A1);
  k_prep_w1<<<96, 256, 0, stream>>>(conv1_w, w1h);
  k_prep_w2<<<20736, 256, 0, stream>>>(prim_w, w2h);
  k_transpose_f16<<<320, 256, 0, stream>>>(dec_w1, d1t, 160, 512);
  k_transpose_f16<<<2048, 256, 0, stream>>>(dec_w2, d2t, 512, 1024);
  k_transpose_f16<<<3136, 256, 0, stream>>>(dec_w3, d3t, 1024, 784);

  // conv1 + relu (WMMA, 16x64 wave tiles)
  k_conv1_gemm<<<3200, wblk, 0, stream>>>(A1, w1h, conv1_b, h_nhwc);
  // primary capsules (WMMA implicit GEMM, 128x64 block tiles, async-LDS staged B)
  k_prim_gemm<<<288, wblk, 0, stream>>>(h_nhwc, w2h, prim_b, p);
  // squash capsule vectors
  k_squash8<<<1152, 256, 0, stream>>>(p);
  // priors einsum
  k_priors<<<184320, 256, 0, stream>>>(p, route_w, priors);
  // dynamic routing (3 iters)
  k_routing<<<2560, 128, 0, stream>>>(priors, vout);
  // classes + mask
  k_classes<<<1, 256, 0, stream>>>(vout, out_classes, masked);
  // decoder (WMMA)
  k_fc_gemm<<<64, wblk, 0, stream>>>(masked, d1t, dec_b1, 512, 160, 0, h1, nullptr);
  k_fc_gemm<<<128, wblk, 0, stream>>>(h1, d2t, dec_b2, 1024, 512, 0, h2, nullptr);
  k_fc_gemm<<<98, wblk, 0, stream>>>(h2, d3t, dec_b3, 784, 1024, 1, nullptr, out_recon);

  (void)in_sizes; (void)n_in; (void)out_size; (void)ws_size;
}
